// AttentionBlock_51453708206791
// MI455X (gfx1250) — matrix operand order repaired and checked by reading, not compiled, not run on hardware
//
#include <hip/hip_runtime.h>
#include <hip/hip_bf16.h>

typedef _Float16 h8  __attribute__((ext_vector_type(8)));
typedef _Float16 h16 __attribute__((ext_vector_type(16)));
typedef float    f8  __attribute__((ext_vector_type(8)));

#define BATCH  8
#define CDIM   512
#define NH     8
#define DH     64          // head dim
#define NSP    1024        // H*W tokens
#define NGRP   32
#define CG     16          // channels per group
#define QKV_O  1536

union F16x16 { h16 v; h8 p[2]; };

__device__ __forceinline__ f8 wmma_f16(h16 a, h16 b, f8 c) {
  // D = A(16x32 f16) * B(32x16 f16) + C(16x16 f32)
  return __builtin_amdgcn_wmma_f32_16x16x32_f16(false, a, false, b, (short)0, c,
                                                false, false);
}

// A-fragment 16x32 f16. rowptr = &A[row][k0], kb = (lane>>4)*8.
// Lane layout: halves 0..15 = A[m, 2*kb..2*kb+15] (contiguous, matches B)
__device__ __forceinline__ h16 afrag(const _Float16* rowptr, int kb) {
  F16x16 f;
  f.v = *(const h16*)(rowptr + 2 * kb);
  return f.v;
}

// async memory -> LDS copy of 16 bytes (CDNA5 GLOBAL_LOAD_ASYNC_TO_LDS_B128)
__device__ __forceinline__ void async_copy_b128(void* lds_dst, const void* gsrc) {
  unsigned lds_off = (unsigned)(uintptr_t)lds_dst;   // low 32 bits give the LDS address
  asm volatile("global_load_async_to_lds_b128 %0, %1, off"
               :: "v"(lds_off), "v"(gsrc) : "memory");
}

__device__ __forceinline__ void wait_async0() {
  asm volatile("s_wait_asynccnt 0x0" ::: "memory");
}

// ---------------------------------------------------------------------------
// Kernel 1: fp32 -> f16 weight conversion
// ---------------------------------------------------------------------------
__global__ void cvt_weights(const float* __restrict__ qkv_w,
                            const float* __restrict__ proj_w,
                            _Float16* __restrict__ qw16,
                            _Float16* __restrict__ pw16) {
  int i = blockIdx.x * 256 + threadIdx.x;
  if (i < QKV_O * CDIM) qw16[i] = (_Float16)qkv_w[i];
  if (i < CDIM * CDIM)  pw16[i] = (_Float16)proj_w[i];
}

// ---------------------------------------------------------------------------
// Kernel 2: GroupNorm(32, 512) -> hn_t[B][N][C] (token-major, f16)
// ---------------------------------------------------------------------------
__global__ void groupnorm_kernel(const float* __restrict__ x,
                                 const float* __restrict__ gw,
                                 const float* __restrict__ gb,
                                 _Float16* __restrict__ hn_t) {
  const int b = blockIdx.x / NGRP;
  const int g = blockIdx.x % NGRP;
  const float* xg = x + ((size_t)b * CDIM + (size_t)g * CG) * NSP;

  float s = 0.f, ss = 0.f;
  for (int i = threadIdx.x; i < CG * NSP; i += 256) {
    float v = xg[i];
    s += v; ss += v * v;
  }
  __shared__ float r0[256];
  __shared__ float r1[256];
  r0[threadIdx.x] = s; r1[threadIdx.x] = ss;
  __syncthreads();
  for (int st = 128; st > 0; st >>= 1) {
    if (threadIdx.x < st) {
      r0[threadIdx.x] += r0[threadIdx.x + st];
      r1[threadIdx.x] += r1[threadIdx.x + st];
    }
    __syncthreads();
  }
  const float inv_n = 1.f / (float)(CG * NSP);
  const float mean = r0[0] * inv_n;
  const float var  = r1[0] * inv_n - mean * mean;
  const float rstd = rsqrtf(var + 1e-5f);

  for (int i = threadIdx.x; i < CG * NSP; i += 256) {
    int cl = i & (CG - 1);
    int n  = i >> 4;
    int c  = g * CG + cl;
    float v = (xg[(size_t)cl * NSP + n] - mean) * rstd * gw[c] + gb[c];
    hn_t[((size_t)b * NSP + n) * CDIM + c] = (_Float16)v;
  }
}

// ---------------------------------------------------------------------------
// Kernel 3: Q/K projection, transposed orientation, 64x32 wave tile.
// out_t[n][o] = sum_c hn_t[b][n][c] * qkv_w[o][c],  o in [0,1024)
// ---------------------------------------------------------------------------
__global__ void qk_gemm_t(const _Float16* __restrict__ hn_t,
                          const _Float16* __restrict__ w16,
                          const float* __restrict__ qkv_b,
                          _Float16* __restrict__ qbuf,
                          _Float16* __restrict__ kbuf) {
  const int lane = threadIdx.x & 31;
  const int wid  = threadIdx.x >> 5;
  const int w    = blockIdx.x * 8 + wid;       // 4096 waves
  const int b    = w >> 9;
  const int rem  = w & 511;                    // 16 token-tiles x 32 oc-tiles
  const int tn   = (rem >> 5) << 6;            // token tile (64 rows)
  const int to   = (rem & 31) << 5;            // out-channel tile (32 cols)
  const int kb   = (lane >> 4) * 8;
  const int kb2  = (lane >> 4) * 16;
  const int lm   = lane & 15;

  const _Float16* Ab = hn_t + (size_t)b * NSP * CDIM;
  f8 acc[4][2] = {};

  for (int c0 = 0; c0 < CDIM; c0 += 32) {
    h16 a[4], bb[2];
#pragma unroll
    for (int mi = 0; mi < 4; ++mi)
      a[mi] = afrag(Ab + (size_t)(tn + mi * 16 + lm) * CDIM + c0, kb);
#pragma unroll
    for (int ni = 0; ni < 2; ++ni)
      bb[ni] = *(const h16*)(w16 + (size_t)(to + ni * 16 + lm) * CDIM + c0 + kb2);
#pragma unroll
    for (int mi = 0; mi < 4; ++mi)
#pragma unroll
      for (int ni = 0; ni < 2; ++ni)
        acc[mi][ni] = wmma_f16(a[mi], bb[ni], acc[mi][ni]);
  }

  const float scale = 0.125f;                  // d^-0.5, d=64
#pragma unroll
  for (int mi = 0; mi < 4; ++mi) {
#pragma unroll
    for (int ni = 0; ni < 2; ++ni) {
      int oc = to + ni * 16 + lm;              // column = out channel
#pragma unroll
      for (int j = 0; j < 8; ++j) {
        int row = tn + mi * 16 + j + 8 * (lane >> 4);   // token
        float val = acc[mi][ni][j] + qkv_b[oc];
        if (oc < 512) {
          int h = oc >> 6, dl = oc & 63;
          qbuf[(((size_t)b * NH + h) * NSP + row) * DH + dl] =
              (_Float16)(val * scale);
        } else {
          int oc2 = oc - 512;
          int h = oc2 >> 6, dl = oc2 & 63;
          kbuf[(((size_t)b * NH + h) * NSP + row) * DH + dl] = (_Float16)val;
        }
      }
    }
  }
}

// ---------------------------------------------------------------------------
// Kernel 4: V projection, normal orientation, 64x32 wave tile.
// out[o][n] = sum_c qkv_w[1024+o][c] * hn_t[b][n][c]  -> v[b][h][d][n]
// ---------------------------------------------------------------------------
__global__ void v_gemm(const _Float16* __restrict__ hn_t,
                       const _Float16* __restrict__ w16,
                       const float* __restrict__ qkv_b,
                       _Float16* __restrict__ vbuf) {
  const int lane = threadIdx.x & 31;
  const int wid  = threadIdx.x >> 5;
  const int w    = blockIdx.x * 8 + wid;       // 2048 waves
  const int b    = w >> 8;
  const int rem  = w & 255;                    // 8 oc-tiles x 32 token-tiles
  const int to   = (rem >> 5) << 6;            // out-channel tile (64)
  const int tn   = (rem & 31) << 5;            // token tile (32)
  const int kb   = (lane >> 4) * 8;
  const int kb2  = (lane >> 4) * 16;
  const int lm   = lane & 15;

  const _Float16* Ab = w16 + (size_t)1024 * CDIM;   // V rows of qkv_w
  const _Float16* Bb = hn_t + (size_t)b * NSP * CDIM;
  f8 acc[4][2] = {};

  for (int c0 = 0; c0 < CDIM; c0 += 32) {
    h16 a[4], bb[2];
#pragma unroll
    for (int mi = 0; mi < 4; ++mi)
      a[mi] = afrag(Ab + (size_t)(to + mi * 16 + lm) * CDIM + c0, kb);
#pragma unroll
    for (int ni = 0; ni < 2; ++ni)
      bb[ni] = *(const h16*)(Bb + (size_t)(tn + ni * 16 + lm) * CDIM + c0 + kb2);
#pragma unroll
    for (int mi = 0; mi < 4; ++mi)
#pragma unroll
      for (int ni = 0; ni < 2; ++ni)
        acc[mi][ni] = wmma_f16(a[mi], bb[ni], acc[mi][ni]);
  }

#pragma unroll
  for (int mi = 0; mi < 4; ++mi) {
#pragma unroll
    for (int ni = 0; ni < 2; ++ni) {
      int n = tn + ni * 16 + lm;
#pragma unroll
      for (int j = 0; j < 8; ++j) {
        int m = to + mi * 16 + j + 8 * (lane >> 4);   // channel in [0,512)
        int h = m >> 6, dl = m & 63;
        float val = acc[mi][ni][j] + qkv_b[1024 + m];
        vbuf[(((size_t)b * NH + h) * DH + dl) * NSP + n] = (_Float16)val;
      }
    }
  }
}

// ---------------------------------------------------------------------------
// Kernel 5: flash attention.  One block = (b,h, 128 query rows), 8 waves,
// each wave owns 16 rows. K/V chunks staged to LDS with async-to-LDS copies.
// ---------------------------------------------------------------------------
__global__ __launch_bounds__(256) void attn_kernel(
    const _Float16* __restrict__ qbuf, const _Float16* __restrict__ kbuf,
    const _Float16* __restrict__ vbuf, _Float16* __restrict__ o_t) {
  __shared__ __attribute__((aligned(32))) _Float16 ldsK[32 * DH];   // [key][d]
  __shared__ __attribute__((aligned(32))) _Float16 ldsV[DH * 32];   // [d][key]
  __shared__ __attribute__((aligned(32))) _Float16 ldsP[8 * 16 * 32];

  const int lane = threadIdx.x & 31;
  const int wid  = threadIdx.x >> 5;
  const int bh   = blockIdx.x >> 3;            // 64 (b,h) pairs
  const int tile = blockIdx.x & 7;
  const int b    = bh >> 3;
  const int h    = bh & 7;
  const int row0 = tile * 128 + wid * 16;
  const int kb   = (lane >> 4) * 8;
  const int kb2  = (lane >> 4) * 16;
  const int lm   = lane & 15;

  const _Float16* Qb = qbuf + ((size_t)bh) * NSP * DH;
  const _Float16* Kb = kbuf + ((size_t)bh) * NSP * DH;
  const _Float16* Vb = vbuf + ((size_t)bh) * DH * NSP;
  _Float16* Pw = ldsP + wid * (16 * 32);

  // preload this wave's 16 Q rows as two A-frags (d = 0..31, 32..63)
  h16 qa0 = afrag(Qb + (size_t)(row0 + lm) * DH + 0, kb);
  h16 qa1 = afrag(Qb + (size_t)(row0 + lm) * DH + 32, kb);

  f8 o_acc[4] = {};
  float m_run[8], l_run[8];
#pragma unroll
  for (int j = 0; j < 8; ++j) { m_run[j] = -1e30f; l_run[j] = 0.f; }

  for (int kc = 0; kc < NSP; kc += 32) {
    __syncthreads();
    {   // async stage of K(32x64) and V(64x32) chunks: memory -> LDS directly
      int t = threadIdx.x;
      int key = t >> 3, seg = (t & 7) * 8;
      async_copy_b128(ldsK + key * DH + seg,
                      Kb + (size_t)(kc + key) * DH + seg);
      int dr = t >> 2, s2 = (t & 3) * 8;
      async_copy_b128(ldsV + dr * 32 + s2,
                      Vb + (size_t)dr * NSP + kc + s2);
      if (kc + 32 < NSP)
        __builtin_prefetch(Kb + (size_t)(kc + 32 + (t & 31)) * DH, 0, 1);
    }
    wait_async0();
    __syncthreads();

    // S tiles: 16 rows x 32 keys, f32
    f8 s0 = {}, s1 = {};
    {
      const _Float16* kr0 = ldsK + (0 + lm) * DH;
      const _Float16* kr1 = ldsK + (16 + lm) * DH;
      s0 = wmma_f16(qa0, *(const h16*)(kr0 + kb2), s0);
      s0 = wmma_f16(qa1, *(const h16*)(kr0 + 32 + kb2), s0);
      s1 = wmma_f16(qa0, *(const h16*)(kr1 + kb2), s1);
      s1 = wmma_f16(qa1, *(const h16*)(kr1 + 32 + kb2), s1);
    }

    // online softmax per row (rows split across j and lane halves)
#pragma unroll
    for (int j = 0; j < 8; ++j) {
      float rm = fmaxf(s0[j], s1[j]);
      rm = fmaxf(rm, __shfl_xor(rm, 1));
      rm = fmaxf(rm, __shfl_xor(rm, 2));
      rm = fmaxf(rm, __shfl_xor(rm, 4));
      rm = fmaxf(rm, __shfl_xor(rm, 8));
      float mnew = fmaxf(m_run[j], rm);
      float corr = __expf(m_run[j] - mnew);
      float p0 = __expf(s0[j] - mnew);
      float p1 = __expf(s1[j] - mnew);
      float ps = p0 + p1;
      ps += __shfl_xor(ps, 1);
      ps += __shfl_xor(ps, 2);
      ps += __shfl_xor(ps, 4);
      ps += __shfl_xor(ps, 8);
      l_run[j] = l_run[j] * corr + ps;
      m_run[j] = mnew;
#pragma unroll
      for (int dt = 0; dt < 4; ++dt) o_acc[dt][j] *= corr;
      int m = j + 8 * (lane >> 4);
      Pw[m * 32 + lm]      = (_Float16)p0;   // D-layout -> A-layout via LDS
      Pw[m * 32 + 16 + lm] = (_Float16)p1;
    }

    // wave-local LDS RAW: DS ops are in-order, wait for stores explicitly
    asm volatile("s_wait_dscnt 0" ::: "memory");
    h16 pa = afrag(Pw + (size_t)lm * 32, kb);

#pragma unroll
    for (int dt = 0; dt < 4; ++dt) {
      h16 bv = *(const h16*)(ldsV + (dt * 16 + lm) * 32 + kb2);
      o_acc[dt] = wmma_f16(pa, bv, o_acc[dt]);
    }
  }

  // normalize and store o_t[b][row][h*64 + d]
#pragma unroll
  for (int j = 0; j < 8; ++j) {
    float inv = 1.f / l_run[j];
    int row = row0 + j + 8 * (lane >> 4);
    size_t base = ((size_t)b * NSP + row) * CDIM + h * DH;
#pragma unroll
    for (int dt = 0; dt < 4; ++dt)
      o_t[base + dt * 16 + lm] = (_Float16)(o_acc[dt][j] * inv);
  }
}

// ---------------------------------------------------------------------------
// Kernel 6: output projection + bias + residual, fp32 out, 64x32 wave tile.
// ---------------------------------------------------------------------------
__global__ void proj_gemm(const _Float16* __restrict__ o_t,
                          const _Float16* __restrict__ pw16,
                          const float* __restrict__ proj_b,
                          const float* __restrict__ x,
                          float* __restrict__ out) {
  const int lane = threadIdx.x & 31;
  const int wid  = threadIdx.x >> 5;
  const int w    = blockIdx.x * 8 + wid;       // 2048 waves
  const int b    = w >> 8;
  const int rem  = w & 255;
  const int to   = (rem >> 5) << 6;            // out-channel tile (64)
  const int tn   = (rem & 31) << 5;            // token tile (32)
  const int kb   = (lane >> 4) * 8;
  const int kb2  = (lane >> 4) * 16;
  const int lm   = lane & 15;

  const _Float16* Bb = o_t + (size_t)b * NSP * CDIM;
  f8 acc[4][2] = {};

  for (int c0 = 0; c0 < CDIM; c0 += 32) {
    h16 a[4], bb[2];
#pragma unroll
    for (int mi = 0; mi < 4; ++mi)
      a[mi] = afrag(pw16 + (size_t)(to + mi * 16 + lm) * CDIM + c0, kb);
#pragma unroll
    for (int ni = 0; ni < 2; ++ni)
      bb[ni] = *(const h16*)(Bb + (size_t)(tn + ni * 16 + lm) * CDIM + c0 + kb2);
#pragma unroll
    for (int mi = 0; mi < 4; ++mi)
#pragma unroll
      for (int ni = 0; ni < 2; ++ni)
        acc[mi][ni] = wmma_f16(a[mi], bb[ni], acc[mi][ni]);
  }

#pragma unroll
  for (int mi = 0; mi < 4; ++mi) {
#pragma unroll
    for (int ni = 0; ni < 2; ++ni) {
      int n = tn + ni * 16 + lm;
#pragma unroll
      for (int j = 0; j < 8; ++j) {
        int oc = to + mi * 16 + j + 8 * (lane >> 4);
        size_t idx = ((size_t)b * CDIM + oc) * NSP + n;
        out[idx] = acc[mi][ni][j] + proj_b[oc] + x[idx];
      }
    }
  }
}

// ---------------------------------------------------------------------------
extern "C" void kernel_launch(void* const* d_in, const int* in_sizes, int n_in,
                              void* d_out, int out_size, void* d_ws,
                              size_t ws_size, hipStream_t stream) {
  const float* x      = (const float*)d_in[0];
  const float* gn_w   = (const float*)d_in[1];
  const float* gn_b   = (const float*)d_in[2];
  const float* qkv_w  = (const float*)d_in[3];
  const float* qkv_b  = (const float*)d_in[4];
  const float* proj_w = (const float*)d_in[5];
  const float* proj_b = (const float*)d_in[6];
  float* out          = (float*)d_out;

  // workspace layout (f16): hn_t, q, k, v, o_t, qkv_w16, proj_w16  (~44 MB)
  _Float16* p = (_Float16*)d_ws;
  const size_t TOK = (size_t)BATCH * NSP * CDIM;   // 4M halves each
  _Float16* hn_t = p;            p += TOK;
  _Float16* qbuf = p;            p += TOK;
  _Float16* kbuf = p;            p += TOK;
  _Float16* vbuf = p;            p += TOK;
  _Float16* o_t  = p;            p += TOK;
  _Float16* qw16 = p;            p += (size_t)QKV_O * CDIM;
  _Float16* pw16 = p;            p += (size_t)CDIM * CDIM;

  cvt_weights<<<(QKV_O * CDIM + 255) / 256, 256, 0, stream>>>(qkv_w, proj_w,
                                                              qw16, pw16);
  groupnorm_kernel<<<BATCH * NGRP, 256, 0, stream>>>(x, gn_w, gn_b, hn_t);
  qk_gemm_t<<<512, 256, 0, stream>>>(hn_t, qw16, qkv_b, qbuf, kbuf);
  v_gemm<<<256, 256, 0, stream>>>(hn_t, qw16, qkv_b, vbuf);
  attn_kernel<<<512, 256, 0, stream>>>(qbuf, kbuf, vbuf, o_t);
  proj_gemm<<<256, 256, 0, stream>>>(o_t, pw16, proj_b, x, out);
}
